// Attention_24842090840792
// MI455X (gfx1250) — matrix operand order repaired and checked by reading, not compiled, not run on hardware
//
#include <hip/hip_runtime.h>

// ---- problem constants (match reference) ----
#define B_ 4
#define N_ 4096
#define H_ 256
#define MASK_SCALE 1000000.0f

typedef __attribute__((ext_vector_type(16))) __bf16         v16bf;
typedef __attribute__((ext_vector_type(8)))  __bf16         v8bf;
typedef __attribute__((ext_vector_type(8)))  float          v8f;
typedef __attribute__((ext_vector_type(4)))  float          v4f;
typedef __attribute__((ext_vector_type(8)))  unsigned short v8us;

__device__ inline v8f wmma_bf16(v16bf a, v16bf b, v8f c) {
    // (neg_a, A, neg_b, B, c_mod, C, reuse_a, reuse_b)
    return __builtin_amdgcn_wmma_f32_16x16x32_bf16(false, a, false, b, (short)0, c,
                                                   false, false);
}

__device__ inline v8bf ld_v8bf(const __bf16* p) {
    return *reinterpret_cast<const v8bf*>(p);
}

__device__ inline v16bf cat16(v8bf lo, v8bf hi) {
    v16bf r;
#pragma unroll
    for (int i = 0; i < 8; ++i) { r[i] = lo[i]; r[i + 8] = hi[i]; }
    return r;
}

__device__ inline v8f zero8() {
    v8f z;
#pragma unroll
    for (int i = 0; i < 8; ++i) z[i] = 0.0f;
    return z;
}

__device__ inline unsigned short bf16_bits(float x) {
    __bf16 h = (__bf16)x;
    return __builtin_bit_cast(unsigned short, h);
}

// ---------------------------------------------------------------------------
// Kernel 1: convert k -> bf16 row-major [B,N,H]; v -> bf16 transposed [B,H,N]
// (transposed V makes the WMMA B-operand of P*V contiguous along keys)
// ---------------------------------------------------------------------------
__global__ void cvt_kv_kernel(const float* __restrict__ k,
                              const float* __restrict__ v,
                              __bf16* __restrict__ kbf,
                              __bf16* __restrict__ vt) {
    size_t idx = (size_t)blockIdx.x * blockDim.x + threadIdx.x;  // exact grid
    kbf[idx] = (__bf16)k[idx];
    int h = (int)(idx % H_);
    int n = (int)((idx / H_) % N_);
    int b = (int)(idx / ((size_t)H_ * N_));
    vt[((size_t)b * H_ + h) * N_ + n] = (__bf16)v[idx];
}

// ---------------------------------------------------------------------------
// Kernel 2: qp = q @ W^T + bias, output bf16 row-major [B*N, H]
// One wave per 16x16 output tile; K = 256 = 8 x wmma_f32_16x16x32_bf16
// ---------------------------------------------------------------------------
__global__ void __launch_bounds__(32)
proj_kernel(const float* __restrict__ q, const float* __restrict__ W,
            const float* __restrict__ bias, __bf16* __restrict__ qp) {
    int lane = threadIdx.x & 31;
    int r  = lane & 15;
    int hf = lane >> 4;
    int mtile = blockIdx.x;   // 16 rows of B*N
    int otile = blockIdx.y;   // 16 output columns of H

    int row  = mtile * 16 + r;
    int ocol = otile * 16 + r;

    const float* qrow = q + (size_t)row * H_;
    const float* wrow = W + (size_t)ocol * H_;   // B[k=h][n=o] = W[o][h]

    v8f acc = zero8();
#pragma unroll
    for (int c = 0; c < 8; ++c) {
        // A fragment: lane=row, K = hf*16 + i contiguous (this code's order, matched by B)
        const float* a0 = qrow + c * 32 + hf * 16;
        const float* a1 = a0 + 8;
        v4f qa = ((const v4f*)a0)[0], qb = ((const v4f*)a0)[1];
        v4f qc = ((const v4f*)a1)[0], qd = ((const v4f*)a1)[1];
        v16bf A;
#pragma unroll
        for (int i = 0; i < 4; ++i) {
            A[i]      = (__bf16)qa[i];  A[i + 4]  = (__bf16)qb[i];
            A[i + 8]  = (__bf16)qc[i];  A[i + 12] = (__bf16)qd[i];
        }
        // B fragment: lane=column, K = hf*16 + i contiguous (same order as A)
        const float* b0 = wrow + c * 32 + hf * 16;
        v4f w0 = ((const v4f*)b0)[0], w1 = ((const v4f*)b0)[1];
        v4f w2 = ((const v4f*)b0)[2], w3 = ((const v4f*)b0)[3];
        v16bf Bm;
#pragma unroll
        for (int i = 0; i < 4; ++i) {
            Bm[i]      = (__bf16)w0[i];  Bm[i + 4]  = (__bf16)w1[i];
            Bm[i + 8]  = (__bf16)w2[i];  Bm[i + 12] = (__bf16)w3[i];
        }
        acc = wmma_bf16(A, Bm, acc);
    }

    float bv = bias[ocol];
#pragma unroll
    for (int j = 0; j < 8; ++j) {   // C layout: VGPR j = row j + hf*8, col = r
        int orow = mtile * 16 + j + hf * 8;
        qp[(size_t)orow * H_ + otile * 16 + r] = (__bf16)(acc[j] + bv);
    }
}

// ---------------------------------------------------------------------------
// Kernel 3: flash attention. One wave owns 16 query rows x full H=256 output.
// Loop over 32-key blocks: S = QK^T (16 wmma), mask, online softmax,
// P transposed via LDS into A-layout, O += P*V (16 wmma).
// ---------------------------------------------------------------------------
__global__ void __launch_bounds__(32)
flash_kernel(const __bf16* __restrict__ qp, const __bf16* __restrict__ kbf,
             const __bf16* __restrict__ vt, const float* __restrict__ mask,
             float* __restrict__ out) {
    __shared__ unsigned short ldsP[16 * 32];   // 1KB: P tile 16 rows x 32 keys (bf16)

    int lane = threadIdx.x & 31;
    int r  = lane & 15;
    int hf = lane >> 4;

    int wg    = blockIdx.x;
    int b     = wg / (N_ / 16);
    int qtile = wg % (N_ / 16);
    int qbase = qtile * 16;

    // Q tile 16x256 as 8 A-fragments, kept in registers for the whole sweep
    v16bf aq[8];
    const __bf16* qrow = qp + ((size_t)b * N_ + qbase + r) * H_;
#pragma unroll
    for (int c = 0; c < 8; ++c) {
        v8bf lo = ld_v8bf(qrow + c * 32 + hf * 16);
        v8bf hi = ld_v8bf(qrow + c * 32 + hf * 16 + 8);
        aq[c] = cat16(lo, hi);
    }

    v8f acc[16];                               // O tile 16x256 (16 column tiles)
#pragma unroll
    for (int t = 0; t < 16; ++t) acc[t] = zero8();

    float m_i[8], l_i[8];                      // per-row stats (rows j + hf*8)
#pragma unroll
    for (int j = 0; j < 8; ++j) { m_i[j] = -3.0e38f; l_i[j] = 0.0f; }

    const __bf16* kbase = kbf + (size_t)b * N_ * H_;
    const __bf16* vbase = vt + (size_t)b * H_ * N_;
    const float*  mbase = mask + ((size_t)b * N_ + qbase) * N_;

    for (int jb = 0; jb < N_; jb += 32) {
        // ---- S = Q K^T : two 16x16 score tiles over K=256 ----
        v8f s0 = zero8(), s1 = zero8();
        const __bf16* k0 = kbase + (size_t)(jb + r) * H_;        // B col = key
        const __bf16* k1 = kbase + (size_t)(jb + 16 + r) * H_;
#pragma unroll
        for (int c = 0; c < 8; ++c) {
            v16bf bk0 = cat16(ld_v8bf(k0 + c * 32 + hf * 16),
                              ld_v8bf(k0 + c * 32 + hf * 16 + 8));
            s0 = wmma_bf16(aq[c], bk0, s0);
            v16bf bk1 = cat16(ld_v8bf(k1 + c * 32 + hf * 16),
                              ld_v8bf(k1 + c * 32 + hf * 16 + 8));
            s1 = wmma_bf16(aq[c], bk1, s1);
        }

        // ---- mask + online softmax (C layout: lane owns col r, rows j+hf*8) ----
        float p0[8], p1[8];
#pragma unroll
        for (int j = 0; j < 8; ++j) {
            const float* mrow = mbase + (size_t)(j + hf * 8) * N_ + jb;
            float sm0 = s0[j] - MASK_SCALE * (1.0f - mrow[r]);
            float sm1 = s1[j] - MASK_SCALE * (1.0f - mrow[16 + r]);
            float mx = fmaxf(sm0, sm1);                 // reduce across 16 cols
            mx = fmaxf(mx, __shfl_xor(mx, 1, 32));
            mx = fmaxf(mx, __shfl_xor(mx, 2, 32));
            mx = fmaxf(mx, __shfl_xor(mx, 4, 32));
            mx = fmaxf(mx, __shfl_xor(mx, 8, 32));
            float mnew  = fmaxf(m_i[j], mx);
            float scale = __expf(m_i[j] - mnew);
            float e0 = __expf(sm0 - mnew);
            float e1 = __expf(sm1 - mnew);
            float rs = e0 + e1;
            rs += __shfl_xor(rs, 1, 32);
            rs += __shfl_xor(rs, 2, 32);
            rs += __shfl_xor(rs, 4, 32);
            rs += __shfl_xor(rs, 8, 32);
            l_i[j] = l_i[j] * scale + rs;
            m_i[j] = mnew;
            p0[j] = e0;
            p1[j] = e1;
#pragma unroll
            for (int t = 0; t < 16; ++t) acc[t][j] *= scale;
        }

        // ---- transpose P (C layout -> A layout) through LDS ----
        __syncthreads();
#pragma unroll
        for (int j = 0; j < 8; ++j) {
            int row = j + hf * 8;
            ldsP[row * 32 + r]      = bf16_bits(p0[j]);
            ldsP[row * 32 + 16 + r] = bf16_bits(p1[j]);
        }
        __syncthreads();
        v8us u0 = *reinterpret_cast<const v8us*>(&ldsP[r * 32 + hf * 16]);
        v8us u1 = *reinterpret_cast<const v8us*>(&ldsP[r * 32 + hf * 16 + 8]);
        v16bf aP = cat16(__builtin_bit_cast(v8bf, u0), __builtin_bit_cast(v8bf, u1));

        // ---- O += P * V : 16 column tiles, K = 32 keys ----
#pragma unroll
        for (int t = 0; t < 16; ++t) {
            const __bf16* vp = vbase + (size_t)(t * 16 + r) * N_ + jb + hf * 16;
            v16bf bv = cat16(ld_v8bf(vp), ld_v8bf(vp + 8));
            acc[t] = wmma_bf16(aP, bv, acc[t]);
        }
    }

    // ---- epilogue: normalize and store fp32 output ----
    float* obase = out + ((size_t)b * N_ + qbase) * H_;
#pragma unroll
    for (int t = 0; t < 16; ++t) {
#pragma unroll
        for (int j = 0; j < 8; ++j) {
            int row = j + hf * 8;
            obase[(size_t)row * H_ + t * 16 + r] = acc[t][j] / l_i[j];
        }
    }
}

// ---------------------------------------------------------------------------
extern "C" void kernel_launch(void* const* d_in, const int* in_sizes, int n_in,
                              void* d_out, int out_size, void* d_ws, size_t ws_size,
                              hipStream_t stream) {
    const float* q    = (const float*)d_in[0];
    const float* k    = (const float*)d_in[1];
    const float* v    = (const float*)d_in[2];
    const float* mask = (const float*)d_in[3];
    const float* W    = (const float*)d_in[4];
    const float* bias = (const float*)d_in[5];
    float* out = (float*)d_out;

    const size_t elems = (size_t)B_ * N_ * H_;   // 4M elements
    __bf16* qp  = (__bf16*)d_ws;                 // 8 MB
    __bf16* kbf = qp + elems;                    // 8 MB
    __bf16* vt  = kbf + elems;                   // 8 MB

    cvt_kv_kernel<<<dim3((unsigned)(elems / 256)), dim3(256), 0, stream>>>(k, v, kbf, vt);
    proj_kernel<<<dim3((B_ * N_) / 16, H_ / 16), dim3(32), 0, stream>>>(q, W, bias, qp);
    flash_kernel<<<dim3(B_ * (N_ / 16)), dim3(32), 0, stream>>>(qp, kbf, vt, mask, out);
}
